// MultiHeadAttention_35837207118195
// MI455X (gfx1250) — matrix operand order repaired and checked by reading, not compiled, not run on hardware
//
#include <hip/hip_runtime.h>
#include <hip/hip_bf16.h>

#define BB  2
#define TT  4096
#define DD  512
#define HH  8
#define DHH 64

typedef __attribute__((ext_vector_type(16))) __bf16 v16bf;
typedef __attribute__((ext_vector_type(8)))  float  v8f;
typedef __attribute__((ext_vector_type(8)))  unsigned short u16x8;

struct alignas(16) F4 { float x, y, z, w; };
struct F4x2 { F4 a, b; };
struct US4  { unsigned short a, b, c, d; };

static __device__ __forceinline__ unsigned short f2bf(float f) {
  unsigned u = __builtin_bit_cast(unsigned, f);
  return (unsigned short)((u + 0x7FFFu + ((u >> 16) & 1u)) >> 16);  // RNE
}

static __device__ __forceinline__ v16bf make_frag(F4 lo, F4 hi) {
  F4x2 t{lo, hi};
  return __builtin_bit_cast(v16bf, t);
}

// LDS byte offset of a shared-memory pointer (flat LDS aperture keeps the
// offset in addr[31:0] per the CDNA5 aperture mapping).
static __device__ __forceinline__ unsigned lds_off(const void* p) {
  return (unsigned)(unsigned long long)(size_t)p;
}

// Async DMA: 16 bytes global -> LDS, tracked by ASYNCcnt (no VGPR staging).
// INST_OFFSET is added to BOTH the global and LDS addresses (ISA 08 §4.4).
static __device__ __forceinline__ void async_b128(unsigned lds, const void* gp) {
  unsigned long long ga = (unsigned long long)(size_t)gp;
  asm volatile("global_load_async_to_lds_b128 %0, %1, off"
               :: "v"(lds), "v"(ga) : "memory");
}
static __device__ __forceinline__ void async_b128_o16(unsigned lds, const void* gp) {
  unsigned long long ga = (unsigned long long)(size_t)gp;
  asm volatile("global_load_async_to_lds_b128 %0, %1, off offset:16"
               :: "v"(lds), "v"(ga) : "memory");
}
static __device__ __forceinline__ void wait_async() {
  asm volatile("s_wait_asynccnt 0" ::: "memory");
}

// ---------------------------------------------------------------- cast f32->bf16
__global__ void cast4_kernel(const float* __restrict__ in,
                             unsigned short* __restrict__ out, int n4) {
  int i = blockIdx.x * blockDim.x + threadIdx.x;
  if (i < n4) {
    F4 f = reinterpret_cast<const F4*>(in)[i];
    US4 r{f2bf(f.x), f2bf(f.y), f2bf(f.z), f2bf(f.w)};
    reinterpret_cast<US4*>(out)[i] = r;
  }
}

// ---------------------------------------------------------------- bf16 WMMA GEMM
// C[M,N] = A[M,K] (bf16) x W[N,K]^T (bf16) + bias[N]
// Block tile 128x64, 8 waves each owning 32x32; K staged in 32-chunks via
// async global->LDS DMA (ASYNCcnt), fragments via ds_load_b128.
template <bool OUT_BF16, bool HEAD>
__global__ __launch_bounds__(256) void gemm_kernel(
    const unsigned short* __restrict__ A,
    const unsigned short* __restrict__ W,
    const float* __restrict__ bias,
    void* __restrict__ outp, int M, int N, int K) {
  __shared__ alignas(16) unsigned short Als[128 * 40];  // stride 40 (80B, bank-spread)
  __shared__ alignas(16) unsigned short Bls[64 * 40];

  const int tid  = threadIdx.x;
  const int lane = tid & 31;
  const int half = lane >> 4;
  const int ln   = lane & 15;
  const int wave = tid >> 5;
  const int wm   = wave >> 1;   // 0..3 (M)
  const int wn   = wave & 1;    // 0..1 (N)
  const int mBase = blockIdx.x * 128;
  const int nBase = blockIdx.y * 64;

  v8f acc[2][2] = {};

  const int arow = tid >> 1;          // 0..127
  const int acs  = (tid & 1) * 16;    // 0,16
  const int brow = tid >> 2;          // 0..63
  const int bcs  = (tid & 3) * 8;     // 0,8,16,24

  const unsigned ldsA = lds_off(Als + arow * 40 + acs);
  const unsigned ldsB = lds_off(Bls + brow * 40 + bcs);

  for (int k0 = 0; k0 < K; k0 += 32) {
    // async stage A (128x32) and B=W rows (64x32): global -> LDS, no VGPRs
    async_b128    (ldsA, A + (size_t)(mBase + arow) * K + k0 + acs);
    async_b128_o16(ldsA, A + (size_t)(mBase + arow) * K + k0 + acs);
    async_b128    (ldsB, W + (size_t)(nBase + brow) * K + k0 + bcs);
    if (k0 + 32 < K)  // pull next A chunk toward L2 (global_prefetch_b8)
      __builtin_prefetch(A + (size_t)(mBase + arow) * K + k0 + 32 + acs, 0, 1);
    wait_async();
    __syncthreads();

#pragma unroll
    for (int i = 0; i < 2; ++i) {
      const int mrow = wm * 32 + i * 16 + ln;
      const unsigned short* ap = Als + mrow * 40 + half * 8;  // A layout: K halves interleave
      F4 alo = *reinterpret_cast<const F4*>(ap);
      F4 ahi = *reinterpret_cast<const F4*>(ap + 16);
      v16bf av = make_frag(alo, ahi);
#pragma unroll
      for (int j = 0; j < 2; ++j) {
        const int nrow = wn * 32 + j * 16 + ln;
        const unsigned short* bp = Bls + nrow * 40 + half * 8;  // B layout: K halves interleave
        F4 blo = *reinterpret_cast<const F4*>(bp);
        F4 bhi = *reinterpret_cast<const F4*>(bp + 16);
        v16bf bv = make_frag(blo, bhi);
        acc[i][j] = __builtin_amdgcn_wmma_f32_16x16x32_bf16(
            false, av, false, bv, (short)0, acc[i][j], false, false);
      }
    }
    __syncthreads();
  }

  // epilogue: bias add, optional head-major scatter [B,H,T,DH]
#pragma unroll
  for (int j = 0; j < 2; ++j) {
    const int col = nBase + wn * 32 + j * 16 + ln;
    const float bval = bias[col];
#pragma unroll
    for (int i = 0; i < 2; ++i) {
#pragma unroll
      for (int r = 0; r < 8; ++r) {
        const int row = mBase + wm * 32 + i * 16 + r + half * 8;
        float val = acc[i][j][r] + bval;
        size_t idx;
        if (HEAD) {
          int bb = row >> 12, t = row & (TT - 1);
          int hh = col >> 6,  dh = col & 63;
          idx = (((size_t)(bb * HH + hh)) * TT + t) * DHH + dh;
        } else {
          idx = (size_t)row * N + col;
        }
        if (OUT_BF16) reinterpret_cast<unsigned short*>(outp)[idx] = f2bf(val);
        else          reinterpret_cast<float*>(outp)[idx] = val;
      }
    }
  }
}

// ---------------------------------------------------------------- flash attention
// One block = one (b,h) x 128 query rows; 8 waves x 16 rows each.
// Keys streamed in 64-wide blocks (K tile via async global->LDS DMA);
// online softmax; scores scaled by 1/DH (reference divides by sqrt(DH) twice).
// mask==0 rows -> constant scores -> uniform softmax (matches NEG fill).
__global__ __launch_bounds__(256) void attn_kernel(
    const unsigned short* __restrict__ Q,   // [B,H,T,DH] bf16
    const unsigned short* __restrict__ Km,  // [B,H,T,DH] bf16
    const unsigned short* __restrict__ Vm,  // [B,H,T,DH] bf16
    const float* __restrict__ mask,         // [B,T]
    unsigned short* __restrict__ O) {       // [B,T,D] bf16
  __shared__ alignas(16) unsigned short Kt[64 * 72];      // keys row-major
  __shared__ alignas(16) unsigned short Vt[64 * 72];      // V transposed: [dh][key]
  __shared__ alignas(16) unsigned short Pw[8][16 * 72];   // per-wave P scratch

  const int tid  = threadIdx.x;
  const int lane = tid & 31;
  const int half = lane >> 4;
  const int ln   = lane & 15;
  const int wave = tid >> 5;
  const int qt = blockIdx.x;   // 0..31
  const int bh = blockIdx.y;   // 0..15
  const int b = bh >> 3, h = bh & 7;
  const float invDH = 1.0f / (float)DHH;

  const size_t headBase = (size_t)bh * TT * DHH;
  const int qRow0 = qt * 128 + wave * 16;

  // Q fragments (two K-chunks over DH=64), A-operand layout, straight from global
  v16bf Qf[2];
  {
    const unsigned short* qp = Q + headBase + (size_t)(qRow0 + ln) * DHH;
#pragma unroll
    for (int c = 0; c < 2; ++c) {
      const unsigned short* p = qp + c * 32 + half * 8;
      F4 lo = *reinterpret_cast<const F4*>(p);
      F4 hi = *reinterpret_cast<const F4*>(p + 16);
      Qf[c] = make_frag(lo, hi);
    }
  }

  // per-row score multiplier (0 for masked rows -> uniform softmax)
  float mr[8];
#pragma unroll
  for (int r = 0; r < 8; ++r) {
    float mv = mask[b * TT + qRow0 + r + half * 8];
    mr[r] = (mv == 0.0f) ? 0.0f : invDH;
  }

  float rowM[8], rowS[8];
#pragma unroll
  for (int r = 0; r < 8; ++r) { rowM[r] = -3.0e38f; rowS[r] = 0.0f; }
  v8f Ot[4] = {};

  const int skey = tid >> 2;        // 0..63
  const int scs  = (tid & 3) * 16;  // 0,16,32,48
  const unsigned ldsK = lds_off(Kt + skey * 72 + scs);

  for (int s0 = 0; s0 < TT; s0 += 64) {
    // stage K tile via async DMA (row-major), V tile transposed by hand
    {
      const unsigned short* gk = Km + headBase + (size_t)(s0 + skey) * DHH + scs;
      async_b128    (ldsK, gk);
      async_b128_o16(ldsK, gk);

      const F4* gv = reinterpret_cast<const F4*>(Vm + headBase + (size_t)(s0 + skey) * DHH + scs);
      F4 v0 = gv[0], v1 = gv[1];
      u16x8 e0 = __builtin_bit_cast(u16x8, v0);
      u16x8 e1 = __builtin_bit_cast(u16x8, v1);
#pragma unroll
      for (int i = 0; i < 8; ++i) {
        Vt[(scs + i) * 72 + skey]     = e0[i];
        Vt[(scs + 8 + i) * 72 + skey] = e1[i];
      }
      wait_async();
    }
    __syncthreads();

    // S = Q x K^T : four 16x16 tiles, 2 chained WMMAs over DH
    v8f S[4];
#pragma unroll
    for (int j = 0; j < 4; ++j) {
      v8f s = {};
#pragma unroll
      for (int c = 0; c < 2; ++c) {
        const unsigned short* kp = Kt + (j * 16 + ln) * 72 + c * 32 + half * 8;
        F4 blo = *reinterpret_cast<const F4*>(kp);
        F4 bhi = *reinterpret_cast<const F4*>(kp + 16);
        v16bf bv = make_frag(blo, bhi);
        s = __builtin_amdgcn_wmma_f32_16x16x32_bf16(false, Qf[c], false, bv, (short)0, s, false, false);
      }
      S[j] = s;
    }

    // online softmax (rows live in VGPR index + lane-half; reduce across 16 lanes)
    float corr[8];
#pragma unroll
    for (int r = 0; r < 8; ++r) {
      float s0v = S[0][r] * mr[r], s1v = S[1][r] * mr[r];
      float s2v = S[2][r] * mr[r], s3v = S[3][r] * mr[r];
      float smax = fmaxf(fmaxf(s0v, s1v), fmaxf(s2v, s3v));
#pragma unroll
      for (int m = 1; m < 16; m <<= 1) smax = fmaxf(smax, __shfl_xor(smax, m, 32));
      float rmNew = fmaxf(rowM[r], smax);
      float c = __expf(rowM[r] - rmNew);
      float p0 = __expf(s0v - rmNew), p1 = __expf(s1v - rmNew);
      float p2 = __expf(s2v - rmNew), p3 = __expf(s3v - rmNew);
      S[0][r] = p0; S[1][r] = p1; S[2][r] = p2; S[3][r] = p3;
      float ps = p0 + p1 + p2 + p3;
#pragma unroll
      for (int m = 1; m < 16; m <<= 1) ps += __shfl_xor(ps, m, 32);
      rowS[r] = rowS[r] * c + ps;
      rowM[r] = rmNew;
      corr[r] = c;
    }
#pragma unroll
    for (int jt = 0; jt < 4; ++jt)
#pragma unroll
      for (int r = 0; r < 8; ++r) Ot[jt][r] *= corr[r];

    // transpose P (C-layout -> A-layout) through per-wave LDS scratch
#pragma unroll
    for (int j = 0; j < 4; ++j)
#pragma unroll
      for (int r = 0; r < 8; ++r)
        Pw[wave][(r + half * 8) * 72 + j * 16 + ln] = f2bf(S[j][r]);

    asm volatile("s_wait_dscnt 0" ::: "memory");  // in-wave cross-lane LDS visibility
    __builtin_amdgcn_wave_barrier();

    // O += P x V
#pragma unroll
    for (int jt = 0; jt < 4; ++jt) {
#pragma unroll
      for (int kc = 0; kc < 2; ++kc) {
        const unsigned short* pp = &Pw[wave][ln * 72 + kc * 32 + half * 8];
        F4 alo = *reinterpret_cast<const F4*>(pp);
        F4 ahi = *reinterpret_cast<const F4*>(pp + 16);
        v16bf av = make_frag(alo, ahi);
        const unsigned short* vp = Vt + (jt * 16 + ln) * 72 + kc * 32 + half * 8;
        F4 blo = *reinterpret_cast<const F4*>(vp);
        F4 bhi = *reinterpret_cast<const F4*>(vp + 16);
        v16bf bv = make_frag(blo, bhi);
        Ot[jt] = __builtin_amdgcn_wmma_f32_16x16x32_bf16(
            false, av, false, bv, (short)0, Ot[jt], false, false);
      }
    }
    __syncthreads();
  }

  // normalize + store head-interleaved [B,T,D] so final GEMM reads row-major
#pragma unroll
  for (int jt = 0; jt < 4; ++jt) {
#pragma unroll
    for (int r = 0; r < 8; ++r) {
      int row = qRow0 + r + half * 8;
      int dh  = jt * 16 + ln;
      float val = Ot[jt][r] / rowS[r];
      O[((size_t)(b * TT + row)) * DD + h * 64 + dh] = f2bf(val);
    }
  }
}

// ---------------------------------------------------------------- launch
extern "C" void kernel_launch(void* const* d_in, const int* in_sizes, int n_in,
                              void* d_out, int out_size, void* d_ws, size_t ws_size,
                              hipStream_t stream) {
  (void)in_sizes; (void)n_in; (void)out_size; (void)ws_size;
  const float* q   = (const float*)d_in[0];
  const float* msk = (const float*)d_in[1];
  const float* Wq  = (const float*)d_in[2];
  const float* bq  = (const float*)d_in[3];
  const float* Wk  = (const float*)d_in[4];
  const float* bk  = (const float*)d_in[5];
  const float* Wv  = (const float*)d_in[6];
  const float* bv  = (const float*)d_in[7];
  const float* Wo  = (const float*)d_in[8];
  const float* bo  = (const float*)d_in[9];
  float* out = (float*)d_out;

  const size_t XE = (size_t)BB * TT * DD;  // 4,194,304 elems
  const size_t WE = (size_t)DD * DD;       //   262,144 elems
  char* w = (char*)d_ws;
  unsigned short* Xb  = (unsigned short*)w; w += XE * 2;  // query bf16 (reused as O later)
  unsigned short* Wqb = (unsigned short*)w; w += WE * 2;
  unsigned short* Wkb = (unsigned short*)w; w += WE * 2;
  unsigned short* Wvb = (unsigned short*)w; w += WE * 2;
  unsigned short* Wob = (unsigned short*)w; w += WE * 2;
  unsigned short* Qb  = (unsigned short*)w; w += XE * 2;
  unsigned short* Kb  = (unsigned short*)w; w += XE * 2;
  unsigned short* Vb  = (unsigned short*)w; w += XE * 2;
  unsigned short* Ob  = Xb;  // alias: Xb dead after QKV GEMMs

  const int n4 = (int)(XE / 4), w4 = (int)(WE / 4);
  cast4_kernel<<<(n4 + 255) / 256, 256, 0, stream>>>(q,  Xb,  n4);
  cast4_kernel<<<(w4 + 255) / 256, 256, 0, stream>>>(Wq, Wqb, w4);
  cast4_kernel<<<(w4 + 255) / 256, 256, 0, stream>>>(Wk, Wkb, w4);
  cast4_kernel<<<(w4 + 255) / 256, 256, 0, stream>>>(Wv, Wvb, w4);
  cast4_kernel<<<(w4 + 255) / 256, 256, 0, stream>>>(Wo, Wob, w4);

  dim3 gg((BB * TT) / 128, DD / 64);  // 64 x 8
  gemm_kernel<true,  true ><<<gg, 256, 0, stream>>>(Xb, Wqb, bq, Qb, BB * TT, DD, DD);
  gemm_kernel<true,  true ><<<gg, 256, 0, stream>>>(Xb, Wkb, bk, Kb, BB * TT, DD, DD);
  gemm_kernel<true,  true ><<<gg, 256, 0, stream>>>(Xb, Wvb, bv, Vb, BB * TT, DD, DD);

  attn_kernel<<<dim3(TT / 128, BB * HH), 256, 0, stream>>>(Qb, Kb, Vb, msk, Ob);

  gemm_kernel<false, false><<<gg, 256, 0, stream>>>(Ob, Wob, bo, (void*)out, BB * TT, DD, DD);
}
